// PDNetPostProcessor_51402168598856
// MI455X (gfx1250) — compile-verified, matrix operand order checked by reading, not run on hardware
//
#include <hip/hip_runtime.h>
#include <hip/hip_bf16.h>
#include <math.h>

#ifndef __has_builtin
#define __has_builtin(x) 0
#endif

#if defined(__HIP_DEVICE_COMPILE__) && __has_builtin(__builtin_amdgcn_global_load_async_to_lds_b128)
#define HAVE_ASYNC_LDS 1
#else
#define HAVE_ASYNC_LDS 0
#endif

typedef __attribute__((ext_vector_type(16))) _Float16 v16h;
typedef __attribute__((ext_vector_type(8)))  float    v8f;
typedef int v4i_vs __attribute__((vector_size(16)));

#define NIMG 8
#define NCLS 80
#define KSEL 1000
#define CAND 3000            // 3 levels * 1000
#define TMAX 188             // ceil(3000/16)
#define MASK_PER_IMG 564000  // 3000 * 188 u16 words (worst case)

__device__ __forceinline__ unsigned mono_key(float f) {
  unsigned b = __float_as_uint(f);
  unsigned m = (unsigned)((int)b >> 31) | 0x80000000u;
  return b ^ m;  // monotonic ascending key
}

__device__ __forceinline__ unsigned ballot32(bool p) {
#if __has_builtin(__builtin_amdgcn_ballot_w32)
  return __builtin_amdgcn_ballot_w32(p);
#else
  return (unsigned)__ballot((int)p);
#endif
}

__device__ __forceinline__ void wait_async0() {
#if __has_builtin(__builtin_amdgcn_s_wait_asynccnt)
  __builtin_amdgcn_s_wait_asynccnt(0);
#else
  asm volatile("s_wait_asynccnt 0" ::: "memory");
#endif
}

// ---------------------------------------------------------------- init
__global__ void init_kernel(unsigned* hist, unsigned* state, int* classCount,
                            int* keptFlag) {
  int i = blockIdx.x * 256 + threadIdx.x;
  if (i < 24 * 256) hist[i] = 0;
  if (i < 24) {
    state[i * 4 + 0] = 0;      // prefix
    state[i * 4 + 1] = KSEL;   // remaining
    state[i * 4 + 2] = 0;      // cntGT
    state[i * 4 + 3] = 0;      // cntEQ
  }
  if (i < NIMG * NCLS) classCount[i] = 0;
  if (i < NIMG * CAND) keptFlag[i] = 0;
}

// ------------------------------------------------- radix-select histogram
// Block handles one 4KB chunk of one image's logits (staged via async->LDS).
__global__ void hist_kernel(const float* __restrict__ cls, int elems, int lvl,
                            unsigned* __restrict__ hist,
                            const unsigned* __restrict__ state, int pass) {
  __shared__ __align__(16) float stage[1024];
  __shared__ unsigned lhist[256];
  const int t = threadIdx.x;     // 0..255
  const int img = blockIdx.y;
  const int i0 = blockIdx.x * 1024 + 4 * t;
  lhist[t] = 0;
  const float* src = cls + (long)img * elems;
  const bool inb = (i0 < elems);  // elems % 4 == 0 for all levels

#if HAVE_ASYNC_LDS
  if (inb) {
    typedef __attribute__((address_space(1))) v4i_vs* as1_v4i;
    typedef __attribute__((address_space(3))) v4i_vs* as3_v4i;
    __builtin_amdgcn_global_load_async_to_lds_b128(
        (as1_v4i)(unsigned long long)(const void*)(src + i0),
        (as3_v4i)(unsigned)(unsigned long long)(void*)(&stage[4 * t]),
        0, 0);
  }
  wait_async0();
#else
  if (inb) *(float4*)&stage[4 * t] = *(const float4*)(src + i0);
#endif
  __syncthreads();

  const unsigned pfx = state[(img * 3 + lvl) * 4 + 0];
  if (inb) {
#pragma unroll
    for (int e = 0; e < 4; ++e) {
      unsigned k = mono_key(stage[4 * t + e]);
      bool ok = (pass == 3) ? true : ((k >> ((pass + 1) * 8)) == pfx);
      if (ok) atomicAdd(&lhist[(k >> (8 * pass)) & 255u], 1u);
    }
  }
  __syncthreads();
  if (lhist[t]) atomicAdd(&hist[((img * 3 + lvl) << 8) + t], lhist[t]);
}

// ---------------------------------------------------- radix-select reduce
__global__ void select_kernel(unsigned* hist, unsigned* state) {
  int t = threadIdx.x;
  if (t < 24) {
    unsigned* h = hist + (t << 8);
    unsigned pfx = state[t * 4 + 0], rem = state[t * 4 + 1];
    unsigned cum = 0, newrem = rem;
    int chosen = 0;
    for (int b = 255; b >= 0; --b) {
      unsigned c = h[b];
      if (cum + c >= rem) { chosen = b; newrem = rem - cum; break; }
      cum += c;
    }
    state[t * 4 + 0] = (pfx << 8) | (unsigned)chosen;
    state[t * 4 + 1] = newrem;
  }
  __syncthreads();
  for (int i = t; i < 24 * 256; i += 256) hist[i] = 0;
}

// ------------------------------------------------ compaction + box decode
__global__ void compact_kernel(const float* __restrict__ cls,
                               const float* __restrict__ bbox,
                               const int* __restrict__ imgsz, int elems, int HW,
                               int lvl, float stride, unsigned* state,
                               float* candScore, float4* candBox,
                               int* candLabel, int* candValid) {
  const int img = blockIdx.y;
  const int i = blockIdx.x * 256 + threadIdx.x;
  if (i >= elems) return;
  const float logit = cls[(long)img * elems + i];
  const unsigned k = mono_key(logit);
  unsigned* st = state + (img * 3 + lvl) * 4;
  const unsigned T = st[0], Q = st[1];
  int pos = -1;
  if (k > T) {
    pos = (int)atomicAdd(&st[2], 1u);
  } else if (k == T) {
    unsigned e = atomicAdd(&st[3], 1u);
    if (e < Q) pos = (int)(KSEL - Q + e);
  }
  if (pos < 0 || pos >= KSEL) return;

  const int c = i / HW, hw = i % HW;
  const float score = 1.f / (1.f + expf(-logit));
  const float* bb = bbox + (long)img * 4 * HW;
  float b0 = bb[hw] * stride, b1 = bb[HW + hw] * stride;
  float b2 = bb[2 * HW + hw] * stride, b3 = bb[3 * HW + hw] * stride;
  const float hI = (float)imgsz[img * 2 + 0], wI = (float)imgsz[img * 2 + 1];
  float m0 = fminf(b0, hI), m1 = fminf(b1, wI);
  float m2 = fminf(b2, hI), m3 = fminf(b3, wI);
  float x1 = fminf(fmaxf(m1, 0.f), wI - 1.f);
  float y1 = fminf(fmaxf(m0, 0.f), hI - 1.f);
  float x2 = fminf(fmaxf(fmaxf(m1, m3), 0.f), wI - 1.f);
  float y2 = fminf(fmaxf(fmaxf(m0, m2), 0.f), hI - 1.f);

  const int slot = img * CAND + lvl * KSEL + pos;
  candScore[slot] = score;
  candBox[slot] = make_float4(x1, y1, x2, y2);
  candLabel[slot] = c + 1;
  candValid[slot] = (score > 0.05f) ? 1 : 0;
}

// -------------------------------------------------- per-image bitonic sort
__global__ void sort_kernel(const float* __restrict__ candScore,
                            const int* __restrict__ candValid, int* sortedCand,
                            float* sortedScore) {
  __shared__ float sk[4096];
  __shared__ int sv[4096];
  const int img = blockIdx.x, t = threadIdx.x;  // 1024 threads
  for (int i = t; i < 4096; i += 1024) {
    float kk = -3.f;
    if (i < CAND) {
      int cand = img * CAND + i;
      kk = candValid[cand] ? candScore[cand] : -1.f;  // reference: where(valid,s,-1)
    }
    sk[i] = kk;
    sv[i] = i;
  }
  __syncthreads();
  for (int len = 2; len <= 4096; len <<= 1) {
    for (int j = len >> 1; j > 0; j >>= 1) {
      for (int i = t; i < 4096; i += 1024) {
        int p = i ^ j;
        if (p > i) {
          bool dirDesc = ((i & len) == 0);
          bool sw = dirDesc ? (sk[i] < sk[p]) : (sk[i] > sk[p]);
          if (sw) {
            float tk = sk[i]; sk[i] = sk[p]; sk[p] = tk;
            int tv = sv[i];  sv[i] = sv[p]; sv[p] = tv;
          }
        }
      }
      __syncthreads();
    }
  }
  for (int i = t; i < CAND; i += 1024) {
    sortedCand[img * CAND + i] = sv[i];
    sortedScore[img * CAND + i] = sk[i];
  }
}

// ----------------------------------------------------- per-class counting
__global__ void count_kernel(const int* sortedCand, const int* candValid,
                             const int* candLabel, int* classCount) {
  int p = blockIdx.x * 256 + threadIdx.x;
  if (p >= NIMG * CAND) return;
  int img = p / CAND;
  int cand = img * CAND + sortedCand[p];
  if (candValid[cand]) atomicAdd(&classCount[img * NCLS + candLabel[cand] - 1], 1);
}

__global__ void offset_kernel(const int* classCount, int* segOff, int* maskOff) {
  int img = threadIdx.x;
  if (img >= NIMG) return;
  int run = img * CAND;
  long mrun = 0;
  for (int c = 0; c < NCLS; ++c) {
    int n = classCount[img * NCLS + c];
    segOff[img * NCLS + c] = run;
    maskOff[img * NCLS + c] = (int)mrun;
    run += n;
    mrun += (long)n * ((n + 15) / 16);
  }
}

// ----------------------------------- stable scatter into class segments
__global__ void fill_kernel(const int* segOff, const int* sortedCand,
                            const int* candValid, const int* candLabel,
                            const float4* candBox, int* segPos, float4* segBox,
                            float* segArea) {
  const int img = blockIdx.x / NCLS, c = blockIdx.x % NCLS;
  const int L = threadIdx.x;  // 32
  const int base = segOff[img * NCLS + c];
  int w = 0;
  for (int ch = 0; ch < CAND; ch += 32) {
    int p = ch + L;
    int cand = 0;
    bool pred = false;
    if (p < CAND) {
      cand = img * CAND + sortedCand[img * CAND + p];
      pred = candValid[cand] && (candLabel[cand] == c + 1);
    }
    unsigned m = ballot32(pred);
    if (pred) {
      int off = __popc(m & ((1u << L) - 1u));
      int d = base + w + off;
      segPos[d] = p;
      float4 bx = candBox[cand];
      segBox[d] = bx;
      segArea[d] = (bx.z - bx.x + 1.f) * (bx.w - bx.y + 1.f);
    }
    w += __popc(m);
  }
}

// --------------------- per-class NMS: WMMA area outer-sum + ballot masks
__global__ void nms_kernel(const int* classCount, const int* segOff,
                           const int* maskOff, const int* segPos,
                           const float4* __restrict__ segBox,
                           const float* __restrict__ segArea,
                           unsigned short* maskPool, int* keptFlag) {
  const int img = blockIdx.x / NCLS, c = blockIdx.x % NCLS;
  const int L = threadIdx.x;  // 32 (one wave)
  const int M = classCount[img * NCLS + c];
  if (M <= 0) return;
  const int seg = segOff[img * NCLS + c];
  const int T = (M + 15) >> 4;
  unsigned short* mask = maskPool + (long)img * MASK_PER_IMG + maskOff[img * NCLS + c];

  __shared__ float4 rb[16], cb[16];
  __shared__ float ra[16];
  const float invS = 1.0f / 1024.0f;
  const int n = L & 15;
  const int mAdd = (L >> 4) << 3;

  for (int ti = 0; ti < T; ++ti) {
    __syncthreads();
    if (L < 16) {
      int r = ti * 16 + L;
      rb[L] = (r < M) ? segBox[seg + r] : make_float4(0.f, 0.f, 0.f, 0.f);
      ra[L] = (r < M) ? segArea[seg + r] : 0.f;
    }
    __syncthreads();
    for (int tj = ti; tj < T; ++tj) {
      __syncthreads();
      if (L < 16) {
        int q = tj * 16 + L;
        cb[L] = (q < M) ? segBox[seg + q] : make_float4(0.f, 0.f, 0.f, 0.f);
      }
      __syncthreads();
      // A = [scaled_area_row, 1, 0...], B = [1; scaled_area_col; 0...]
      v16h av, bv;
#pragma unroll
      for (int z = 0; z < 16; ++z) { av[z] = (_Float16)0; bv[z] = (_Float16)0; }
      if (L < 16) {
        av[0] = (_Float16)(ra[L] * invS);
        av[1] = (_Float16)1.0f;
        float ca = (tj * 16 + L < M) ? segArea[seg + tj * 16 + L] : 0.f;
        bv[0] = (_Float16)1.0f;
        bv[1] = (_Float16)(ca * invS);
      }
      v8f acc;
#pragma unroll
      for (int z = 0; z < 8; ++z) acc[z] = 0.f;
      // D[i][j] = area_i + area_j (scaled): rank-2 outer sum via matrix core
      acc = __builtin_amdgcn_wmma_f32_16x16x32_f16(false, av, false, bv,
                                                   (short)0, acc, false, false);
      const float4 cB = cb[n];
#pragma unroll
      for (int r = 0; r < 8; ++r) {
        const int mrow = r + mAdd;
        const int i = ti * 16 + mrow, j = tj * 16 + n;
        const float4 rB = rb[mrow];
        float ww = fminf(rB.z, cB.z) - fmaxf(rB.x, cB.x) + 1.f;
        float hh = fminf(rB.w, cB.w) - fmaxf(rB.y, cB.y) + 1.f;
        float inter = fmaxf(ww, 0.f) * fmaxf(hh, 0.f);
        float sum = acc[r] * 1024.f;  // area_i + area_j
        // IoU > 0.6  <=>  inter > 0.375*(area_i+area_j)
        bool pred = (i < M) && (j < M) && (j > i) && (inter > 0.375f * sum);
        unsigned m32 = ballot32(pred);  // bits0-15: row r, bits16-31: row r+8
        int row0 = ti * 16 + r, row1 = ti * 16 + r + 8;
        if (L == 2 * r && row0 < M)
          mask[row0 * T + tj] = (unsigned short)(m32 & 0xFFFFu);
        if (L == 2 * r + 1 && row1 < M)
          mask[row1 * T + tj] = (unsigned short)(m32 >> 16);
      }
    }
  }

  // serial greedy scan (short: per-class list), single wave
  __shared__ unsigned short rem[192];
  for (int t0 = L; t0 < T; t0 += 32) rem[t0] = 0;
  __syncthreads();
  for (int i = 0; i < M; ++i) {
    unsigned w = rem[i >> 4];
    if (!((w >> (i & 15)) & 1u)) {
      if (L == 0) keptFlag[img * CAND + segPos[seg + i]] = 1;
      for (int t0 = (i >> 4) + L; t0 < T; t0 += 32)
        rem[t0] = (unsigned short)(rem[t0] | mask[i * T + t0]);
    }
    __syncthreads();
  }
}

// ------------------- stable partition (top_k tie semantics) + emit output
__global__ void final_kernel(const int* keptFlag, const int* sortedCand,
                             const float* sortedScore, const float4* candBox,
                             const int* candLabel, float* out) {
  __shared__ int ps[1024];
  const int img = blockIdx.x, t = threadIdx.x;  // 1024 threads
  int f[3], s = 0;
#pragma unroll
  for (int e = 0; e < 3; ++e) {
    int p = 3 * t + e;
    f[e] = (p < CAND) ? keptFlag[img * CAND + p] : 0;
    s += f[e];
  }
  ps[t] = s;
  __syncthreads();
  for (int off = 1; off < 1024; off <<= 1) {
    int v = (t >= off) ? ps[t - off] : 0;
    __syncthreads();
    ps[t] += v;
    __syncthreads();
  }
  const int total = ps[1023];
  int run = ps[t] - s;  // exclusive kept count before 3t
#pragma unroll
  for (int e = 0; e < 3; ++e) {
    int p = 3 * t + e;
    if (p >= CAND) break;
    int kept = f[e];
    int outIdx = kept ? run : (total + (p - run));
    if (outIdx < 100) {
      int cand = img * CAND + sortedCand[img * CAND + p];
      float4 bx = candBox[cand];
      float sc = kept ? sortedScore[img * CAND + p] : -1.0f;
      float* d = out + (img * 100 + outIdx) * 5;
      d[0] = bx.x; d[1] = bx.y; d[2] = bx.z; d[3] = bx.w; d[4] = sc;
      out[4000 + img * 100 + outIdx] = (float)candLabel[cand];
      out[4800 + img * 100 + outIdx] = kept ? 1.0f : 0.0f;
    }
    run += kept;
  }
}

// ================================================================= host
extern "C" void kernel_launch(void* const* d_in, const int* in_sizes, int n_in,
                              void* d_out, int out_size, void* d_ws,
                              size_t ws_size, hipStream_t stream) {
  (void)in_sizes; (void)n_in; (void)out_size; (void)ws_size;
  const int HWs[3] = {100 * 152, 50 * 76, 25 * 38};
  const float strides[3] = {8.f, 16.f, 32.f};
  const float* clsP[3] = {(const float*)d_in[0], (const float*)d_in[2],
                          (const float*)d_in[4]};
  const float* boxP[3] = {(const float*)d_in[1], (const float*)d_in[3],
                          (const float*)d_in[5]};
  const int* imgsz = (const int*)d_in[6];
  float* out = (float*)d_out;

  char* w = (char*)d_ws;
  size_t off = 0;
  auto alloc = [&](size_t bytes) -> void* {
    void* p = w + off;
    off = (off + bytes + 255) & ~(size_t)255;
    return p;
  };
  unsigned* hist = (unsigned*)alloc(24 * 256 * 4);
  unsigned* state = (unsigned*)alloc(24 * 16);
  float* candScore = (float*)alloc(NIMG * CAND * 4);
  float4* candBox = (float4*)alloc(NIMG * CAND * 16);
  int* candLabel = (int*)alloc(NIMG * CAND * 4);
  int* candValid = (int*)alloc(NIMG * CAND * 4);
  int* sortedCand = (int*)alloc(NIMG * CAND * 4);
  float* sortedScore = (float*)alloc(NIMG * CAND * 4);
  int* keptFlag = (int*)alloc(NIMG * CAND * 4);
  int* classCount = (int*)alloc(NIMG * NCLS * 4);
  int* segOff = (int*)alloc(NIMG * NCLS * 4);
  int* maskOff = (int*)alloc(NIMG * NCLS * 4);
  int* segPos = (int*)alloc(NIMG * CAND * 4);
  float4* segBox = (float4*)alloc(NIMG * CAND * 16);
  float* segArea = (float*)alloc(NIMG * CAND * 4);
  unsigned short* maskPool = (unsigned short*)alloc((size_t)NIMG * MASK_PER_IMG * 2);

  init_kernel<<<(NIMG * CAND + 255) / 256, 256, 0, stream>>>(hist, state,
                                                             classCount, keptFlag);
  // 4-pass radix select of top-1000 keys per (image, level), logit domain
  for (int pass = 3; pass >= 0; --pass) {
    for (int l = 0; l < 3; ++l) {
      int elems = NCLS * HWs[l];
      dim3 g((elems + 1023) / 1024, NIMG);
      hist_kernel<<<g, 256, 0, stream>>>(clsP[l], elems, l, hist, state, pass);
    }
    select_kernel<<<1, 256, 0, stream>>>(hist, state);
  }
  // compact + sigmoid + box decode
  for (int l = 0; l < 3; ++l) {
    int elems = NCLS * HWs[l];
    dim3 g((elems + 255) / 256, NIMG);
    compact_kernel<<<g, 256, 0, stream>>>(clsP[l], boxP[l], imgsz, elems, HWs[l],
                                          l, strides[l], state, candScore,
                                          candBox, candLabel, candValid);
  }
  sort_kernel<<<NIMG, 1024, 0, stream>>>(candScore, candValid, sortedCand,
                                         sortedScore);
  count_kernel<<<(NIMG * CAND + 255) / 256, 256, 0, stream>>>(
      sortedCand, candValid, candLabel, classCount);
  offset_kernel<<<1, 32, 0, stream>>>(classCount, segOff, maskOff);
  fill_kernel<<<NIMG * NCLS, 32, 0, stream>>>(segOff, sortedCand, candValid,
                                              candLabel, candBox, segPos,
                                              segBox, segArea);
  nms_kernel<<<NIMG * NCLS, 32, 0, stream>>>(classCount, segOff, maskOff,
                                             segPos, segBox, segArea, maskPool,
                                             keptFlag);
  final_kernel<<<NIMG, 1024, 0, stream>>>(keptFlag, sortedCand, sortedScore,
                                          candBox, candLabel, out);
}
